// Decoder_81020263072228
// MI455X (gfx1250) — matrix operand order repaired and checked by reading, not compiled, not run on hardware
//
#include <hip/hip_runtime.h>
#include <hip/hip_bf16.h>

typedef _Float16 f16;
typedef __attribute__((ext_vector_type(8)))  _Float16 v8h;
typedef __attribute__((ext_vector_type(16))) _Float16 v16h;
typedef __attribute__((ext_vector_type(8)))  float    v8f;
typedef __attribute__((ext_vector_type(4)))  unsigned int u32x4;
typedef __attribute__((ext_vector_type(8)))  int i32x8;
typedef __attribute__((ext_vector_type(4)))  int i32x4;

#define V_   32000
#define E_   512
#define HD_  512
#define H_   1024
#define B_   16
#define T_   128
#define CTX_ 2048
#define G4H  4096

// ---------------- WMMA tile helpers (wave32, 16x16x32 f16 -> f32) ----------------
// A tile: M=16 x K=32. ISA layout: lanes 0-15 -> M=0..15; VGPR0..3 hold
// K = half*8..+7, VGPR4..7 hold K = 16+half*8..+7.
__device__ __forceinline__ v16h load_a_tile(const f16* __restrict__ base, int rowStride,
                                            int k0, int lane) {
    int m = lane & 15, half = lane >> 4;
    const f16* p = base + (size_t)m * rowStride + k0 + half * 8;
    v8h lo = *(const v8h*)p;
    v8h hi = *(const v8h*)(p + 16);
    v16h a;
#pragma unroll
    for (int i = 0; i < 8; ++i) { a[i] = lo[i]; a[i + 8] = hi[i]; }
    return a;
}

// B tile: K=32 x N=16 from row-major weight W[N][K] (B[k][n] = W[n][k]).
// Same K order as A: elements 0..7 hold K = half*8..+7, elements 8..15 hold K = 16+half*8..+7.
__device__ __forceinline__ v16h load_b_tile(const f16* __restrict__ w, int rowStride,
                                            int n0, int k0, int lane) {
    int n = lane & 15, half = lane >> 4;
    const f16* p = w + (size_t)(n0 + n) * rowStride + k0 + half * 8;
    v8h lo = *(const v8h*)p;
    v8h hi = *(const v8h*)(p + 16);
    v16h b;
#pragma unroll
    for (int i = 0; i < 8; ++i) { b[i] = lo[i]; b[8 + i] = hi[i]; }
    return b;
}

__device__ __forceinline__ v8f wmma16(v16h a, v16h b, v8f c) {
    return __builtin_amdgcn_wmma_f32_16x16x32_f16(false, a, false, b, (short)0, c,
                                                  false, false);
}

__device__ __forceinline__ float sigmoidf_(float x) { return 1.0f / (1.0f + __expf(-x)); }

// ---------------- setup kernels ----------------
__global__ void k_f32_to_f16(const float* __restrict__ src, f16* __restrict__ dst, int n) {
    for (int i = blockIdx.x * blockDim.x + threadIdx.x; i < n; i += gridDim.x * blockDim.x)
        dst[i] = (f16)src[i];
}

__global__ void k_add2(const float* __restrict__ a, const float* __restrict__ b,
                       float* __restrict__ o, int n) {
    int i = blockIdx.x * blockDim.x + threadIdx.x;
    if (i < n) o[i] = a[i] + b[i];
}

// ctx[b][s*HD+d] = enc_hidden[s][b][d]
__global__ void k_build_ctx(const float* __restrict__ hid, f16* __restrict__ ctx) {
    int i = blockIdx.x * blockDim.x + threadIdx.x;
    if (i >= B_ * CTX_) return;
    int b = i / CTX_, rem = i % CTX_, s = rem / HD_, d = rem % HD_;
    ctx[i] = (f16)hid[((size_t)s * B_ + b) * HD_ + d];
}

// h0/c0 reshape: state[l][b] = concat(enc[2l,b], enc[2l+1,b])
__global__ void k_init_state(const float* __restrict__ hid, const float* __restrict__ ct,
                             f16* __restrict__ h0b, f16* __restrict__ h1b,
                             float* __restrict__ c01) {
    int i = blockIdx.x * blockDim.x + threadIdx.x;
    if (i >= 2 * B_ * H_) return;
    int l = i / (B_ * H_);
    int rem = i % (B_ * H_);
    int b = rem / H_, j = rem % H_;
    int s = 2 * l + (j >= HD_ ? 1 : 0);
    int d = j & (HD_ - 1);
    size_t src = ((size_t)s * B_ + b) * HD_ + d;
    (l == 0 ? h0b : h1b)[rem] = (f16)hid[src];
    c01[i] = ct[src];
}

// teacher-forcing embedding gather
__global__ void k_embed(const int* __restrict__ tgt, const float* __restrict__ emb,
                        f16* __restrict__ xemb) {
    int i = blockIdx.x * blockDim.x + threadIdx.x;
    if (i >= T_ * B_ * E_) return;
    int t = i / (B_ * E_);
    int rem = i % (B_ * E_);
    int b = rem / E_, e = rem % E_;
    int tok = (t == 0) ? 1 : tgt[b * T_ + (t - 1)];
    xemb[i] = (f16)emb[(size_t)tok * E_ + e];
}

// base0[16][4096] = ctx @ w_ih0[:, E:].T + (b_ih0 + b_hh0)   (M=16,N=4096,K=2048)
__global__ void k_base0(const f16* __restrict__ ctx, const f16* __restrict__ wih0,
                        const float* __restrict__ bi, const float* __restrict__ bh,
                        float* __restrict__ base0) {
    int lane = threadIdx.x & 31;
    int half = lane >> 4, col = lane & 15;
    int wid = (blockIdx.x * blockDim.x + threadIdx.x) >> 5;   // 0..255
    int n0 = wid * 16;
    float bv = bi[n0 + col] + bh[n0 + col];
    v8f acc;
#pragma unroll
    for (int r = 0; r < 8; ++r) acc[r] = bv;
    for (int k0 = 0; k0 < CTX_; k0 += 32) {
        v16h a = load_a_tile(ctx, CTX_, k0, lane);
        acc = wmma16(a, load_b_tile(wih0, 2560, n0, E_ + k0, lane), acc);
    }
#pragma unroll
    for (int r = 0; r < 8; ++r)
        base0[(size_t)(half * 8 + r) * G4H + n0 + col] = acc[r];
}

// ---------------- fused LSTM step (one layer, one timestep) ----------------
// 16 blocks x 4 waves: spread the ~29MB/step weight pull over 16 WGPs (L2
// bandwidth is the step limiter), while each block shares the tiny A operands
// (x_t, h_prev) via LDS. Serial inner loop: ds_load_b128 A + global B -> wmma.
__global__ void k_lstm_step(const f16* __restrict__ x1, int xK,
                            const f16* __restrict__ w1, int w1Stride, int w1Col,
                            const f16* __restrict__ hprev,
                            const f16* __restrict__ w2,
                            const float* __restrict__ cinit, int cinitStride,
                            float* __restrict__ cstate,
                            f16* __restrict__ hnext,
                            f16* __restrict__ outseq) {
    extern __shared__ f16 smem[];          // [16*xK] x, then [16*H_] h
    f16* smX = smem;
    f16* smH = smem + B_ * xK;

    int tid = threadIdx.x;
    int nthr = blockDim.x;
    {
        int nchx = (B_ * xK) >> 3;
        for (int c = tid; c < nchx; c += nthr)
            *(v8h*)(smX + c * 8) = *(const v8h*)(x1 + c * 8);
        int nchh = (B_ * H_) >> 3;
        for (int c = tid; c < nchh; c += nthr)
            *(v8h*)(smH + c * 8) = *(const v8h*)(hprev + c * 8);
    }
    __syncthreads();

    int lane = tid & 31;
    int half = lane >> 4, col = lane & 15;
    int j = (blockIdx.x * nthr + tid) >> 5;                   // 0..63
    int nI = j * 16, nF = H_ + j * 16, nG = 2 * H_ + j * 16, nO = 3 * H_ + j * 16;

    v8f ai, af, ag, ao;
#pragma unroll
    for (int r = 0; r < 8; ++r) {
        const float* cb = cinit + (size_t)(half * 8 + r) * cinitStride;
        ai[r] = cb[nI + col]; af[r] = cb[nF + col];
        ag[r] = cb[nG + col]; ao[r] = cb[nO + col];
    }
    for (int k0 = 0; k0 < xK; k0 += 32) {
        v16h a = load_a_tile(smX, xK, k0, lane);
        ai = wmma16(a, load_b_tile(w1, w1Stride, nI, w1Col + k0, lane), ai);
        af = wmma16(a, load_b_tile(w1, w1Stride, nF, w1Col + k0, lane), af);
        ag = wmma16(a, load_b_tile(w1, w1Stride, nG, w1Col + k0, lane), ag);
        ao = wmma16(a, load_b_tile(w1, w1Stride, nO, w1Col + k0, lane), ao);
    }
    for (int k0 = 0; k0 < H_; k0 += 32) {
        v16h a = load_a_tile(smH, H_, k0, lane);
        ai = wmma16(a, load_b_tile(w2, H_, nI, k0, lane), ai);
        af = wmma16(a, load_b_tile(w2, H_, nF, k0, lane), af);
        ag = wmma16(a, load_b_tile(w2, H_, nG, k0, lane), ag);
        ao = wmma16(a, load_b_tile(w2, H_, nO, k0, lane), ao);
    }
#pragma unroll
    for (int r = 0; r < 8; ++r) {
        int m = half * 8 + r;
        size_t idx = (size_t)m * H_ + j * 16 + col;
        float iv = sigmoidf_(ai[r]);
        float fv = sigmoidf_(af[r]);
        float gv = tanhf(ag[r]);
        float ov = sigmoidf_(ao[r]);
        float cN = fv * cstate[idx] + iv * gv;
        cstate[idx] = cN;
        float hN = ov * tanhf(cN);
        hnext[idx] = (f16)hN;
        if (outseq) outseq[idx] = (f16)tanhf(hN);
    }
}

// ---------------- output projection: logits[b][t][v] ----------------
// Workgroup macrotile 128(M) x 64(N); the 64x64 B k-slice (8KB) is DMA'd into
// LDS by the Tensor Data Mover (one issue per block from wave 0, TENSORcnt
// tracked), shared by all 8 waves -> fc_w L2 traffic cut 8x (~1 GB total).
__global__ void k_fc(const f16* __restrict__ A, const f16* __restrict__ Wv,
                     const float* __restrict__ bias, float* __restrict__ out) {
    __shared__ f16 smB[64 * 64];           // sole static LDS alloc -> offset 0

    int tid = threadIdx.x;
    int lane = tid & 31;
    int half = lane >> 4, col = lane & 15;
    int w = tid >> 5;                      // wave 0..7
    int mb = blockIdx.x / 500;             // 0..15
    int nb = blockIdx.x % 500;
    int n0 = nb * 64;
    int mt = mb * 8 + w;                   // M-tile == timestep t (16 rows = batch)

    // touch smB so it is a live LDS allocation independent of the TDM writes
    for (int c = tid; c < (64 * 64) >> 3; c += 256)
        *(v8h*)(smB + c * 8) = v8h{};

    v8f acc[4];
#pragma unroll
    for (int nt = 0; nt < 4; ++nt) {
        float bv = bias[n0 + nt * 16 + col];
#pragma unroll
        for (int r = 0; r < 8; ++r) acc[nt][r] = bv;
    }

    const f16* arow = A + (size_t)mt * 16 * H_;

    // Tensor DMA descriptor (D#), 2D tile: 64 rows x 64 f16, row stride H_.
    u32x4 g0;
    i32x8 g1;
    i32x4 gz4 = {0, 0, 0, 0};
    i32x8 gz8 = {0, 0, 0, 0, 0, 0, 0, 0};
    g0[0] = 1u;                            // count=1 (valid), user mode
    g0[1] = 0u;                            // lds_addr = 0 (smB at LDS offset 0)
    g1[0] = 0x00010000;                    // data_size=1 (2 bytes); mask=0
    g1[1] = (int)((H_ & 0xFFFF) << 16);    // tensor_dim0 = 1024 (bits 79:48)
    g1[2] = (int)(((unsigned)H_ >> 16) | ((V_ & 0xFFFF) << 16)); // dim0 hi | dim1 lo
    g1[3] = (int)(((unsigned)V_ >> 16) | (64u << 16));           // dim1 hi | tile_dim0=64
    g1[4] = 64;                            // tile_dim1 = 64 rows, tile_dim2 = 0
    g1[5] = H_;                            // tensor_dim0_stride = 1024 (lo 32)
    g1[6] = 0;                             // stride hi | dim1_stride lo
    g1[7] = 0;

    for (int k0 = 0; k0 < H_; k0 += 64) {
        __syncthreads();                   // previous slice fully consumed
        if (tid < 32) {                    // wave 0 issues the tensor DMA
            size_t ga = (size_t)Wv + ((size_t)n0 * H_ + k0) * sizeof(f16);
            g0[2] = (unsigned)ga;
            g0[3] = (unsigned)((ga >> 32) & 0x01FFFFFFu) | 0x80000000u; // type=2
            __builtin_amdgcn_tensor_load_to_lds(g0, g1, gz4, gz4, gz8, 0);
            __builtin_amdgcn_s_wait_tensorcnt(0);
        }
        __syncthreads();
        __builtin_prefetch((const void*)(arow + k0 + 64), 0, 0);
#pragma unroll
        for (int kk = 0; kk < 64; kk += 32) {
            v16h a = load_a_tile(arow, H_, k0 + kk, lane);
#pragma unroll
            for (int nt = 0; nt < 4; ++nt)
                acc[nt] = wmma16(a, load_b_tile(smB, 64, nt * 16, kk, lane), acc[nt]);
        }
    }
#pragma unroll
    for (int nt = 0; nt < 4; ++nt) {
        int v = n0 + nt * 16 + col;
#pragma unroll
        for (int r = 0; r < 8; ++r) {
            int b = half * 8 + r;          // rows of tile mt: t=mt, b=0..15
            out[((size_t)b * T_ + mt) * V_ + v] = acc[nt][r];
        }
    }
}

// ---------------- host orchestration ----------------
extern "C" void kernel_launch(void* const* d_in, const int* in_sizes, int n_in,
                              void* d_out, int out_size, void* d_ws, size_t ws_size,
                              hipStream_t stream) {
    (void)in_sizes; (void)n_in; (void)out_size; (void)ws_size;
    const float* enc_h = (const float*)d_in[0];
    const float* enc_c = (const float*)d_in[1];
    const int*   tgt   = (const int*)d_in[2];
    const float* emb   = (const float*)d_in[3];
    const float* wih0  = (const float*)d_in[4];
    const float* whh0  = (const float*)d_in[5];
    const float* bih0  = (const float*)d_in[6];
    const float* bhh0  = (const float*)d_in[7];
    const float* wih1  = (const float*)d_in[8];
    const float* whh1  = (const float*)d_in[9];
    const float* bih1  = (const float*)d_in[10];
    const float* bhh1  = (const float*)d_in[11];
    const float* fcw   = (const float*)d_in[12];
    const float* fcb   = (const float*)d_in[13];
    float* out = (float*)d_out;

    char* ws = (char*)d_ws;
    size_t off = 0;
    auto alloc = [&](size_t bytes) {
        void* p = ws + off;
        off = (off + bytes + 255) & ~(size_t)255;
        return p;
    };
    f16*   xemb   = (f16*)alloc((size_t)T_ * B_ * E_ * 2);
    f16*   wih0h  = (f16*)alloc((size_t)G4H * 2560 * 2);
    f16*   whh0h  = (f16*)alloc((size_t)G4H * H_ * 2);
    f16*   wih1h  = (f16*)alloc((size_t)G4H * H_ * 2);
    f16*   whh1h  = (f16*)alloc((size_t)G4H * H_ * 2);
    f16*   fcwh   = (f16*)alloc((size_t)V_ * H_ * 2);
    f16*   ctxh   = (f16*)alloc((size_t)B_ * CTX_ * 2);
    float* base0  = (float*)alloc((size_t)B_ * G4H * 4);
    float* base1  = (float*)alloc((size_t)G4H * 4);
    f16*   h0buf0 = (f16*)alloc((size_t)B_ * H_ * 2);
    f16*   h0buf1 = (f16*)alloc((size_t)B_ * H_ * 2);
    f16*   h1buf0 = (f16*)alloc((size_t)B_ * H_ * 2);
    f16*   h1buf1 = (f16*)alloc((size_t)B_ * H_ * 2);
    float* c01    = (float*)alloc((size_t)2 * B_ * H_ * 4);
    f16*   outseq = (f16*)alloc((size_t)T_ * B_ * H_ * 2);
    f16* h0b[2] = {h0buf0, h0buf1};
    f16* h1b[2] = {h1buf0, h1buf1};

    auto conv = [&](const float* s, f16* d, int n) {
        k_f32_to_f16<<<(n + 255) / 256, 256, 0, stream>>>(s, d, n);
    };
    conv(wih0, wih0h, G4H * 2560);
    conv(whh0, whh0h, G4H * H_);
    conv(wih1, wih1h, G4H * H_);
    conv(whh1, whh1h, G4H * H_);
    conv(fcw,  fcwh,  V_ * H_);

    k_embed<<<(T_ * B_ * E_ + 255) / 256, 256, 0, stream>>>(tgt, emb, xemb);
    k_build_ctx<<<(B_ * CTX_ + 255) / 256, 256, 0, stream>>>(enc_h, ctxh);
    k_init_state<<<(2 * B_ * H_ + 255) / 256, 256, 0, stream>>>(enc_h, enc_c,
                                                                h0b[0], h1b[0], c01);
    k_add2<<<(G4H + 255) / 256, 256, 0, stream>>>(bih1, bhh1, base1, G4H);
    k_base0<<<32, 256, 0, stream>>>(ctxh, wih0h, bih0, bhh0, base0);

    size_t sh0 = (size_t)B_ * (E_ + H_) * sizeof(f16);   // 48 KB
    size_t sh1 = (size_t)B_ * (H_ + H_) * sizeof(f16);   // 64 KB
    for (int t = 0; t < T_; ++t) {
        int pi = t & 1, po = pi ^ 1;
        // layer 0: x = [emb_t | ctx]; ctx part pre-folded into base0 -> K = E.
        k_lstm_step<<<16, 128, sh0, stream>>>(
            xemb + (size_t)t * B_ * E_, E_,
            wih0h, 2560, 0,
            h0b[pi], whh0h,
            base0, G4H,
            c01, h0b[po], (f16*)nullptr);
        // layer 1: x = layer-0 hidden of this step.
        k_lstm_step<<<16, 128, sh1, stream>>>(
            h0b[po], H_,
            wih1h, H_, 0,
            h1b[pi], whh1h,
            base1, 0,
            c01 + B_ * H_, h1b[po],
            outseq + (size_t)t * B_ * H_);
    }

    k_fc<<<16 * 500, 256, 0, stream>>>(outseq, fcwh, fcb, out);
}
